// SelfAttention_17291538334034
// MI455X (gfx1250) — matrix operand order repaired and checked by reading, not compiled, not run on hardware
//
#include <hip/hip_runtime.h>
#include <hip/hip_bf16.h>

typedef __attribute__((ext_vector_type(16))) __bf16 v16bf;
typedef __attribute__((ext_vector_type(8)))  __bf16 v8bf;
typedef __attribute__((ext_vector_type(8)))  float  v8f;
typedef __attribute__((ext_vector_type(4)))  unsigned int u32x4;
typedef __attribute__((ext_vector_type(8)))  int i32x8;
typedef __attribute__((ext_vector_type(4)))  int i32x4;

union V16U { v16bf v; v8bf h[2]; };

#if defined(__gfx1250__) && __has_builtin(__builtin_amdgcn_tensor_load_to_lds) && \
    __has_builtin(__builtin_amdgcn_s_wait_tensorcnt)
#define USE_TDM 1
#else
#define USE_TDM 0
#endif

// ---------------- GroupNorm -> bf16 ----------------
__global__ __launch_bounds__(256) void groupnorm_kernel(
    const float* __restrict__ x, const float* __restrict__ gscale,
    const float* __restrict__ gbias, __bf16* __restrict__ out)
{
  constexpr int HW = 4096, C = 512, CG = 16;
  int b = blockIdx.x >> 5;
  int g = blockIdx.x & 31;
  int c0 = g * CG;
  const float* xb = x + (size_t)b * HW * C + c0;
  float s = 0.f, sq = 0.f;
  for (int hw = threadIdx.x; hw < HW; hw += 256) {
    const float4* p = (const float4*)(xb + (size_t)hw * C);
#pragma unroll
    for (int i = 0; i < 4; ++i) {
      float4 v = p[i];
      s  += v.x + v.y + v.z + v.w;
      sq += v.x*v.x + v.y*v.y + v.z*v.z + v.w*v.w;
    }
  }
  __shared__ float red0[256], red1[256];
  red0[threadIdx.x] = s; red1[threadIdx.x] = sq;
  __syncthreads();
  for (int off = 128; off > 0; off >>= 1) {
    if (threadIdx.x < off) {
      red0[threadIdx.x] += red0[threadIdx.x + off];
      red1[threadIdx.x] += red1[threadIdx.x + off];
    }
    __syncthreads();
  }
  float mean = red0[0] * (1.f / 65536.f);
  float var  = red1[0] * (1.f / 65536.f) - mean * mean;
  float rstd = rsqrtf(fmaxf(var, 0.f) + 1e-5f);
  float sc[CG], bi[CG];
#pragma unroll
  for (int i = 0; i < CG; ++i) {
    sc[i] = gscale[c0 + i] * rstd;
    bi[i] = gbias[c0 + i] - mean * sc[i];
  }
  __bf16* ob = out + (size_t)b * HW * C + c0;
  for (int hw = threadIdx.x; hw < HW; hw += 256) {
    const float4* p = (const float4*)(xb + (size_t)hw * C);
    alignas(16) __bf16 tmp[CG];
#pragma unroll
    for (int i = 0; i < 4; ++i) {
      float4 v = p[i];
      tmp[i*4+0] = (__bf16)(v.x * sc[i*4+0] + bi[i*4+0]);
      tmp[i*4+1] = (__bf16)(v.y * sc[i*4+1] + bi[i*4+1]);
      tmp[i*4+2] = (__bf16)(v.z * sc[i*4+2] + bi[i*4+2]);
      tmp[i*4+3] = (__bf16)(v.w * sc[i*4+3] + bi[i*4+3]);
    }
    v8bf* o = (v8bf*)(ob + (size_t)hw * C);
    o[0] = *(v8bf*)&tmp[0];
    o[1] = *(v8bf*)&tmp[8];
  }
}

// ---------------- Softmax (rows of 512, in place) ----------------
__global__ __launch_bounds__(256) void softmax_kernel(float* __restrict__ a)
{
  float* row = a + (size_t)blockIdx.x * 512;
  int t = threadIdx.x;
  float v0 = row[t], v1 = row[t + 256];
  __shared__ float red[256];
  red[t] = fmaxf(v0, v1);
  __syncthreads();
  for (int off = 128; off > 0; off >>= 1) {
    if (t < off) red[t] = fmaxf(red[t], red[t + off]);
    __syncthreads();
  }
  float m = red[0];
  __syncthreads();
  float e0 = __expf(v0 - m), e1 = __expf(v1 - m);
  red[t] = e0 + e1;
  __syncthreads();
  for (int off = 128; off > 0; off >>= 1) {
    if (t < off) red[t] += red[t + off];
    __syncthreads();
  }
  float inv = 1.f / red[0];
  row[t] = e0 * inv;
  row[t + 256] = e1 * inv;
}

// ---------------- Tiled bf16 WMMA GEMM (double-buffered LDS) ----------------
// C[M,N] = A[M,K] * B[K,N] (math view). 256 threads = 8 waves.
// Block tile 128x128x32; waves 4(M) x 2(N); each wave 32x64 = 2x4 wmma accums.
// AMODE: 0 = A bf16 [M,K] row-major (TDM tensor_load_to_lds when available);
//        1 = A bf16 stored as [K,M] (VGPR transpose-stage).
// BMODE: 0 = B fp32 [K,N] (transpose + cvt); 1 = B bf16 [K,N] (transpose);
//        2 = B fp32 stored as [N,K] row-major (direct rows + cvt)  [attn case].
// EPI bits: 1 = add bias[n]; 2 = add resid (fp32, C layout); 4 = bf16 out.
constexpr int BMt = 128, BNt = 128, BKt = 32;
constexpr int LSTR = BKt + 8;   // 40 bf16 = 80 B row stride (16B-multiple)

template <int AMODE, int BMODE, int EPI>
__global__ __launch_bounds__(256) void gemm_kernel(
    const void* __restrict__ Ap, const void* __restrict__ Bp,
    const float* __restrict__ bias, const float* __restrict__ resid,
    void* __restrict__ Cp,
    int M, int N, int K, int lda, int ldb, int ldc,
    long long batchA, long long batchB, long long batchC)
{
  __shared__ __bf16 As[2][BMt * LSTR];
  __shared__ __bf16 Bs[2][BNt * LSTR];

  int tid  = threadIdx.x;
  int lane = tid & 31;
  int wid  = tid >> 5;
  int wm   = wid & 3;     // 0..3 -> 32-row strip
  int wn   = wid >> 2;    // 0..1 -> 64-col strip
  int m0   = blockIdx.y * BMt;
  int n0   = blockIdx.x * BNt;
  int bz   = blockIdx.z;

  const __bf16* Abf = (const __bf16*)Ap + (size_t)bz * batchA;

  v8f acc[2][4] = {};

  // ---- staging helpers ----
  auto stageA = [&](int k0, int buf) {
    if (AMODE == 0) {
#if USE_TDM
      // One TDM per tile, issued by wave 0 only. 2D tile: 32 elems (64 B) x
      // 128 rows, row stride lda elems, hardware pad +16 B per 64 B row ->
      // LDS row stride 80 B == LSTR.
      if (tid < 32) {
        unsigned long long ga =
            (unsigned long long)(const void*)(Abf + (size_t)m0 * lda + k0);
        unsigned ldsa = (unsigned)(unsigned long long)(void*)(&As[buf][0]);
        u32x4 g0;
        g0[0] = 1u;                                   // count=1, user mode
        g0[1] = ldsa;                                 // lds_addr
        g0[2] = (unsigned)(ga & 0xFFFFFFFFull);       // global_addr[31:0]
        g0[3] = (unsigned)((ga >> 32) & 0x01FFFFFFull) | (2u << 30); // type=2
        i32x8 g1;
        g1[0] = (1 << 16)     // data_size = 2 bytes
              | (1 << 20)     // pad_enable
              | (3 << 22)     // pad_interval: every 16 DWORDs (64 B)
              | (3 << 25);    // pad_amount: 4 DWORDs (16 B)
        unsigned td0 = (unsigned)lda;        // tensor_dim0 (>= tile span)
        unsigned td1 = 1u << 20;             // tensor_dim1: large, no OOB
        g1[1] = (int)((td0 & 0xFFFFu) << 16);
        g1[2] = (int)(((td0 >> 16) & 0xFFFFu) | ((td1 & 0xFFFFu) << 16));
        g1[3] = (int)(((td1 >> 16) & 0xFFFFu) | (32u << 16)); // tile_dim0=32
        g1[4] = 128;                                          // tile_dim1=128
        g1[5] = lda;                         // tensor_dim0_stride[31:0]
        g1[6] = 0;
        g1[7] = 0;
        i32x4 gz = {0, 0, 0, 0};
        i32x8 gz8 = {0, 0, 0, 0, 0, 0, 0, 0};
        __builtin_amdgcn_tensor_load_to_lds(g0, g1, gz, gz, gz8, 0);
      }
#else
      int row = tid >> 1;
      int col = (tid & 1) * 16;
      const v8bf* src = (const v8bf*)(Abf + (size_t)(m0 + row) * lda + k0 + col);
      v8bf a0 = src[0], a1 = src[1];
      v8bf* dst = (v8bf*)(&As[buf][row * LSTR + col]);
      dst[0] = a0; dst[1] = a1;
#endif
    } else {
      int kk = tid >> 3;           // 0..31
      int mc = (tid & 7) * 16;     // 0..112
      const v8bf* src = (const v8bf*)(Abf + (size_t)(k0 + kk) * lda + m0 + mc);
      v8bf a0 = src[0], a1 = src[1];
#pragma unroll
      for (int i = 0; i < 8; ++i) As[buf][(mc + i) * LSTR + kk] = a0[i];
#pragma unroll
      for (int i = 0; i < 8; ++i) As[buf][(mc + 8 + i) * LSTR + kk] = a1[i];
    }
  };

  auto stageB = [&](int k0, int buf) {
    if (BMODE == 0) {
      const float* Bf = (const float*)Bp + (size_t)bz * batchB;
      int kk = tid >> 3;
      int nc = (tid & 7) * 16;
      const float4* src = (const float4*)(Bf + (size_t)(k0 + kk) * ldb + n0 + nc);
      alignas(16) float tmp[16];
      *(float4*)&tmp[0]  = src[0];
      *(float4*)&tmp[4]  = src[1];
      *(float4*)&tmp[8]  = src[2];
      *(float4*)&tmp[12] = src[3];
#pragma unroll
      for (int i = 0; i < 16; ++i) Bs[buf][(nc + i) * LSTR + kk] = (__bf16)tmp[i];
    } else if (BMODE == 1) {
      const __bf16* Bb = (const __bf16*)Bp + (size_t)bz * batchB;
      int kk = tid >> 3;
      int nc = (tid & 7) * 16;
      const v8bf* src = (const v8bf*)(Bb + (size_t)(k0 + kk) * ldb + n0 + nc);
      v8bf b0 = src[0], b1 = src[1];
#pragma unroll
      for (int i = 0; i < 8; ++i) Bs[buf][(nc + i) * LSTR + kk] = b0[i];
#pragma unroll
      for (int i = 0; i < 8; ++i) Bs[buf][(nc + 8 + i) * LSTR + kk] = b1[i];
    } else {
      const float* Bf = (const float*)Bp + (size_t)bz * batchB;
      int n  = tid >> 1;
      int kc = (tid & 1) * 16;
      const float4* src = (const float4*)(Bf + (size_t)(n0 + n) * ldb + k0 + kc);
      alignas(16) float tmp[16];
      *(float4*)&tmp[0]  = src[0];
      *(float4*)&tmp[4]  = src[1];
      *(float4*)&tmp[8]  = src[2];
      *(float4*)&tmp[12] = src[3];
      alignas(16) __bf16 bt[16];
#pragma unroll
      for (int i = 0; i < 16; ++i) bt[i] = (__bf16)tmp[i];
      v8bf* dst = (v8bf*)(&Bs[buf][n * LSTR + kc]);
      dst[0] = *(v8bf*)&bt[0];
      dst[1] = *(v8bf*)&bt[8];
    }
  };

  const int nIter = K / BKt;
  stageA(0, 0);
  stageB(0, 0);
#if USE_TDM
  if (AMODE == 0) __builtin_amdgcn_s_wait_tensorcnt(0);
#endif
  __syncthreads();

  for (int it = 0; it < nIter; ++it) {
    int cur = it & 1;
    if (it + 1 < nIter) {          // overlap next-tile staging with WMMAs
      stageA((it + 1) * BKt, cur ^ 1);
      stageB((it + 1) * BKt, cur ^ 1);
    }

    // ---- fragments from buffer `cur` ----
    // A 16x32 bf16 layout: lanes<16 row M=lane, K=0..7 (V0..3) & 16..23
    // (V4..7); lanes>=16 same row, K=8..15 & 24..31.
    v16bf afrag[2], bfrag[4];
    int arow = wm * 32 + (lane & 15);
    int klo  = (lane >> 4) * 8;
#pragma unroll
    for (int mi = 0; mi < 2; ++mi) {
      V16U u;
      u.h[0] = *(const v8bf*)(&As[cur][(arow + mi * 16) * LSTR + klo]);
      u.h[1] = *(const v8bf*)(&As[cur][(arow + mi * 16) * LSTR + 16 + klo]);
      afrag[mi] = u.v;
    }
    // B 32x16 bf16 layout: lanes<16 col N=lane, K=0..7 & 16..23; lanes>=16 K=8..15 & 24..31.
    int bcol = wn * 64 + (lane & 15);
    int bklo = (lane >> 4) * 8;
#pragma unroll
    for (int ni = 0; ni < 4; ++ni) {
      V16U u;
      u.h[0] = *(const v8bf*)(&Bs[cur][(bcol + ni * 16) * LSTR + bklo]);
      u.h[1] = *(const v8bf*)(&Bs[cur][(bcol + ni * 16) * LSTR + 16 + bklo]);
      bfrag[ni] = u.v;
    }
#pragma unroll
    for (int mi = 0; mi < 2; ++mi)
#pragma unroll
      for (int ni = 0; ni < 4; ++ni)
        acc[mi][ni] = __builtin_amdgcn_wmma_f32_16x16x32_bf16(
            false, afrag[mi], false, bfrag[ni], (short)0, acc[mi][ni],
            false, false);

#if USE_TDM
    if (AMODE == 0) __builtin_amdgcn_s_wait_tensorcnt(0);
#endif
    __syncthreads();
  }

  // ---- epilogue: C/D 16x16 f32 layout: lanes<16 N=lane, M=vgpr e;
  //      lanes>=16 N=lane-16, M=8+e ----
  int crow = m0 + wm * 32 + ((lane >> 4) << 3);
  int ccol = n0 + wn * 64 + (lane & 15);
#pragma unroll
  for (int mi = 0; mi < 2; ++mi) {
#pragma unroll
    for (int ni = 0; ni < 4; ++ni) {
      int cc = ccol + ni * 16;
      float bv = (EPI & 1) ? bias[cc] : 0.f;
#pragma unroll
      for (int e = 0; e < 8; ++e) {
        int rr = crow + mi * 16 + e;
        size_t idx = (size_t)bz * batchC + (size_t)rr * ldc + cc;
        float val = acc[mi][ni][e] + bv;
        if (EPI & 2) val += resid[idx];
        if (EPI & 4) ((__bf16*)Cp)[idx] = (__bf16)val;
        else         ((float*)Cp)[idx]  = val;
      }
    }
  }
}

// ---------------- Host orchestration ----------------
extern "C" void kernel_launch(void* const* d_in, const int* in_sizes, int n_in,
                              void* d_out, int out_size, void* d_ws, size_t ws_size,
                              hipStream_t stream) {
  const float* x  = (const float*)d_in[0];
  const float* wq = (const float*)d_in[1];
  const float* bq = (const float*)d_in[2];
  const float* wk = (const float*)d_in[3];
  const float* bk = (const float*)d_in[4];
  const float* wv = (const float*)d_in[5];
  const float* bv = (const float*)d_in[6];
  const float* wp = (const float*)d_in[7];
  const float* bp = (const float*)d_in[8];
  const float* gs = (const float*)d_in[9];
  const float* gb = (const float*)d_in[10];
  float* out = (float*)d_out;

  constexpr int Bn = 8, HW = 4096, C = 512, M = Bn * HW;

  char* ws = (char*)d_ws;
  __bf16* hn = (__bf16*)(ws);                              // 32 MB (later h2)
  __bf16* qb = (__bf16*)(ws + (size_t)32 * 1024 * 1024);   // 32 MB
  __bf16* kb = (__bf16*)(ws + (size_t)64 * 1024 * 1024);   // 32 MB
  __bf16* vb = (__bf16*)(ws + (size_t)96 * 1024 * 1024);   // 32 MB
  float*  at = (float*)(ws + (size_t)128 * 1024 * 1024);   // 8 MB

  dim3 blk(256);

  // 1) GroupNorm -> h_norm (bf16)
  groupnorm_kernel<<<dim3(Bn * 32), blk, 0, stream>>>(x, gs, gb, hn);

  // 2) Q/K/V = h_norm @ w + b  (bf16 out)
  dim3 gq(C / BNt, M / BMt, 1);
  gemm_kernel<0, 0, 5><<<gq, blk, 0, stream>>>(hn, wq, bq, nullptr, qb,
      M, C, C, C, C, C, 0, 0, 0);
  gemm_kernel<0, 0, 5><<<gq, blk, 0, stream>>>(hn, wk, bk, nullptr, kb,
      M, C, C, C, C, C, 0, 0, 0);
  gemm_kernel<0, 0, 5><<<gq, blk, 0, stream>>>(hn, wv, bv, nullptr, vb,
      M, C, C, C, C, C, 0, 0, 0);

  // 3) attn[b] = q[b]^T @ k[b]   (M=N=512, K=4096, fp32 out)
  dim3 ga(C / BNt, C / BMt, Bn);
  gemm_kernel<1, 1, 0><<<ga, blk, 0, stream>>>(qb, kb, nullptr, nullptr, at,
      C, C, HW, C, C, C,
      (long long)HW * C, (long long)HW * C, (long long)C * C);

  // 4) softmax rows (in place)
  softmax_kernel<<<dim3(Bn * C), blk, 0, stream>>>(at);

  // 5) h2[b] = v[b] @ attn[b]^T  -> reuse hn buffer (bf16)
  dim3 ge(C / BNt, HW / BMt, Bn);
  gemm_kernel<0, 2, 4><<<ge, blk, 0, stream>>>(vb, at, nullptr, nullptr, hn,
      HW, C, C, C, C, C,
      (long long)HW * C, (long long)C * C, (long long)HW * C);

  // 6) out = x + h2 @ wp + bp   (fp32 out)
  gemm_kernel<0, 0, 3><<<gq, blk, 0, stream>>>(hn, wp, bp, x, out,
      M, C, C, C, C, C, 0, 0, 0);
}
